// AttentionGCNSW_20005957665497
// MI455X (gfx1250) — matrix operand order repaired and checked by reading, not compiled, not run on hardware
//
#include <hip/hip_runtime.h>
#include <hip/hip_bf16.h>

// ---------------------------------------------------------------------------
// GAT(4 heads x 32) + GRU(128) over T=12 on N=50000 nodes, E=800000 edges/t.
// Dense GEMMs on v_wmma_f32_16x16x32_bf16 (f32 accumulate), with:
//  - B weights pre-packed into fragment-contiguous layout (1x v16bf load/lane)
//  - A fragments loaded as 2 contiguous 16B chunks per lane (b128 loads)
//  - gi/gh accumulators stored fragment-packed (v8f -> 2x b128 stores)
// Edge softmax/aggregation via L2-resident f32 atomics.
// ---------------------------------------------------------------------------

typedef __attribute__((ext_vector_type(16))) __bf16 v16bf;
typedef __attribute__((ext_vector_type(8)))  __bf16 v8bf;
typedef __attribute__((ext_vector_type(8)))  float  v8f;

#define N_NODES   50000
#define T_STEPS   12
#define F_INDIM   128
#define HEADS     4
#define FILT      32
#define D_DIM     128
#define E_EDGES   800000
#define E_TOT     (E_EDGES + N_NODES)
#define NEG_SLOPE 0.2f

// ---------------- WMMA fragment helpers (CDNA5 ISA 7.12.2 layouts) ----------

// A 16x32 bf16 from f32 row-major (row stride lda). Per lane: two contiguous
// 8-element K chunks at k0+8*half and k0+16+8*half.
__device__ inline v16bf load_a_frag_f32(const float* __restrict__ A, int lda,
                                        int m0, int k0, int lane) {
    const int half = lane >> 4;
    const int lr   = lane & 15;
    const float* row = A + (size_t)(m0 + lr) * lda + k0 + 8 * half;
    const float4 c0 = *(const float4*)(row + 0);
    const float4 c1 = *(const float4*)(row + 4);
    const float4 c2 = *(const float4*)(row + 16);
    const float4 c3 = *(const float4*)(row + 20);
    v16bf o;
    o[0]  = (__bf16)c0.x; o[1]  = (__bf16)c0.y; o[2]  = (__bf16)c0.z; o[3]  = (__bf16)c0.w;
    o[4]  = (__bf16)c1.x; o[5]  = (__bf16)c1.y; o[6]  = (__bf16)c1.z; o[7]  = (__bf16)c1.w;
    o[8]  = (__bf16)c2.x; o[9]  = (__bf16)c2.y; o[10] = (__bf16)c2.z; o[11] = (__bf16)c2.w;
    o[12] = (__bf16)c3.x; o[13] = (__bf16)c3.y; o[14] = (__bf16)c3.z; o[15] = (__bf16)c3.w;
    return o;
}

// A 16x32 bf16 from bf16 row-major (row stride lda): 2x 16B vector loads.
__device__ inline v16bf load_a_frag_bf16(const __bf16* __restrict__ A, int lda,
                                         int m0, int k0, int lane) {
    const int half = lane >> 4;
    const int lr   = lane & 15;
    const __bf16* row = A + (size_t)(m0 + lr) * lda + k0 + 8 * half;
    const v8bf lo = *(const v8bf*)(row);
    const v8bf hi = *(const v8bf*)(row + 16);
    v16bf o;
#pragma unroll
    for (int i = 0; i < 8; ++i) { o[i] = lo[i]; o[8 + i] = hi[i]; }
    return o;
}

// B fragment from pre-packed buffer: frag f = 32 lanes x 16 bf16 contiguous.
__device__ inline v16bf load_b_packed(const __bf16* __restrict__ Bp, int f, int lane) {
    return *(const v16bf*)(Bp + ((size_t)f << 9) + (lane << 4));
}

// C/D 16x16 f32 row-major store: lane 0-15 -> col, VGPR r -> row r / r+8.
__device__ inline void store_c_frag(float* __restrict__ C, int ldc,
                                    int m0, int n0, int lane, const v8f& c) {
    const int half = lane >> 4;
    const int lr   = lane & 15;
#pragma unroll
    for (int r = 0; r < 8; ++r)
        C[(size_t)(m0 + r + 8 * half) * ldc + n0 + lr] = c[r];
}

__device__ inline v8f splat8(float v) {
    v8f r;
#pragma unroll
    for (int i = 0; i < 8; ++i) r[i] = v;
    return r;
}

__device__ inline void atomicMaxF32(float* addr, float val) {
    // consistent across sign mix: int-max for >=0, uint-min for <0; init = -inf
    if (val >= 0.0f) atomicMax((int*)addr, __float_as_int(val));
    else             atomicMin((unsigned int*)addr, __float_as_uint(val));
}

// ---------------- weight prep: pack B fragments (bf16) ----------------------
// Fragment f = nt*KT + ks holds the 32x16 tile (k in [ks*32,..), n in [nt*16,..))
// laid out as dst[f*512 + lane*16 + idx] with k = ks*32 + 8*(lane>>4) + idx (idx<8), ks*32 + 16 + 8*(lane>>4) + idx-8 (idx>=8),
// n = nt*16 + (lane&15).  transposed=1 reads src[n*ld + k] (for W_ih/W_hh^T).

__global__ void pack_b(const float* __restrict__ src, __bf16* __restrict__ dst,
                       int KT, int NT, int ld, int transposed) {
    const int tid = blockIdx.x * blockDim.x + threadIdx.x;
    if (tid >= KT * NT * 512) return;
    const int f    = tid >> 9;
    const int rem  = tid & 511;
    const int lane = rem >> 4;
    const int idx  = rem & 15;
    const int ks = f % KT, nt = f / KT;
    const int half = lane >> 4, lr = lane & 15;
    const int k = ks * 32 + (idx < 8 ? 8 * half + idx : 16 + 8 * half + (idx - 8));
    const int n = nt * 16 + lr;
    const float v = transposed ? src[(size_t)n * ld + k] : src[(size_t)k * ld + n];
    dst[tid] = (__bf16)v;
}

// ---------------- GAT projection: h = x_t @ W_gat  (N x 128) ----------------

__global__ void gat_project(const float* __restrict__ x,
                            const __bf16* __restrict__ wgat_p,
                            float* __restrict__ hgat, int t) {
    const int wave = threadIdx.x >> 5;
    const int lane = threadIdx.x & 31;
    const int tile = blockIdx.x * 4 + wave;
    if (tile * 16 >= N_NODES) return;                  // wave-uniform exit
    const int m0 = tile * 16;
    const float* A = x + (size_t)t * F_INDIM;          // row stride T*F_IN

    v16bf a[4];
#pragma unroll
    for (int ks = 0; ks < 4; ++ks)
        a[ks] = load_a_frag_f32(A, T_STEPS * F_INDIM, m0, ks * 32, lane);

#pragma unroll
    for (int nt = 0; nt < 8; ++nt) {
        v8f acc = {};
#pragma unroll
        for (int ks = 0; ks < 4; ++ks) {
            const v16bf b = load_b_packed(wgat_p, nt * 4 + ks, lane);
            acc = __builtin_amdgcn_wmma_f32_16x16x32_bf16(
                false, a[ks], false, b, (short)0, acc, false, false);
        }
        store_c_frag(hgat, D_DIM, m0, nt * 16, lane, acc);
    }
}

// ---- per-(node,head) attention logits sources + init of max/denom ----------

__global__ void compute_al(const float* __restrict__ hgat,
                           const float* __restrict__ a_src,
                           const float* __restrict__ a_dst,
                           float* __restrict__ alsrc, float* __restrict__ aldst,
                           float* __restrict__ mmax, float* __restrict__ denom) {
    const int tid = blockIdx.x * blockDim.x + threadIdx.x;
    if (tid >= N_NODES * HEADS) return;
    const int n = tid >> 2, h = tid & 3;
    const float* hp = hgat + (size_t)n * D_DIM + h * FILT;
    const float* as = a_src + h * FILT;
    const float* ad = a_dst + h * FILT;
    float s = 0.f, d = 0.f;
#pragma unroll
    for (int c = 0; c < FILT; ++c) { s += hp[c] * as[c]; d += hp[c] * ad[c]; }
    alsrc[tid] = s;
    aldst[tid] = d;
    mmax[tid]  = -__builtin_inff();
    denom[tid] = 0.f;
}

// ---- edge pass 1: leaky_relu logits + segment max over dst -----------------

__global__ void edge_logits(const int* __restrict__ ei,
                            const float* __restrict__ alsrc,
                            const float* __restrict__ aldst,
                            float* __restrict__ elog, float* __restrict__ mmax,
                            int t) {
    const int e = blockIdx.x * blockDim.x + threadIdx.x;
    if (e >= E_TOT) return;
    int s, d;
    if (e < E_EDGES) {
        const int* base = ei + (size_t)t * 2 * E_EDGES;
        s = base[e];
        d = base[E_EDGES + e];
    } else {
        s = d = e - E_EDGES;                            // self loop
    }
#pragma unroll
    for (int h = 0; h < HEADS; ++h) {
        float v = alsrc[s * 4 + h] + aldst[d * 4 + h];
        v = v > 0.f ? v : NEG_SLOPE * v;
        elog[(size_t)e * 4 + h] = v;
        atomicMaxF32(&mmax[d * 4 + h], v);
    }
}

// ---- edge pass 2: exp(e - max) + segment sum --------------------------------

__global__ void edge_expsum(const int* __restrict__ ei,
                            float* __restrict__ elog,
                            const float* __restrict__ mmax,
                            float* __restrict__ denom, int t) {
    const int e = blockIdx.x * blockDim.x + threadIdx.x;
    if (e >= E_TOT) return;
    int d;
    if (e < E_EDGES) d = ei[(size_t)t * 2 * E_EDGES + E_EDGES + e];
    else             d = e - E_EDGES;
#pragma unroll
    for (int h = 0; h < HEADS; ++h) {
        const float v = __expf(elog[(size_t)e * 4 + h] - mmax[d * 4 + h]);
        elog[(size_t)e * 4 + h] = v;
        atomicAdd(&denom[d * 4 + h], v);
    }
}

// ---- edge pass 3: alpha-weighted message scatter (wave per edge) -----------

__global__ void edge_message(const int* __restrict__ ei,
                             const float* __restrict__ elog,
                             const float* __restrict__ denom,
                             const float* __restrict__ hgat,
                             float* __restrict__ outacc, int t) {
    const int e = blockIdx.x * 8 + (threadIdx.x >> 5);
    if (e >= E_TOT) return;                             // wave-uniform exit
    const int lane = threadIdx.x & 31;
    int s, d;
    if (e < E_EDGES) {
        const int* base = ei + (size_t)t * 2 * E_EDGES;
        s = base[e];
        d = base[E_EDGES + e];
    } else {
        s = d = e - E_EDGES;
    }
    const int head  = lane >> 3;                        // 4 floats/lane, 32/head
    const float alp = elog[(size_t)e * 4 + head] /
                      (denom[d * 4 + head] + 1e-16f);
    const float4 hv = *(const float4*)(hgat + (size_t)s * D_DIM + lane * 4);
    float* op = outacc + (size_t)d * D_DIM + lane * 4;
    atomicAdd(op + 0, hv.x * alp);
    atomicAdd(op + 1, hv.y * alp);
    atomicAdd(op + 2, hv.z * alp);
    atomicAdd(op + 3, hv.w * alp);
}

// ---- g = relu(aggregated + b_gat), stored bf16 as GRU GEMM input -----------

__global__ void gat_finalize(const float* __restrict__ outacc,
                             const float* __restrict__ bgat,
                             __bf16* __restrict__ gbf) {
    const int tid = blockIdx.x * blockDim.x + threadIdx.x;
    if (tid >= N_NODES * D_DIM) return;
    float v = outacc[tid] + bgat[tid & 127];
    gbf[tid] = (__bf16)fmaxf(v, 0.f);
}

// ---- GRU gate GEMMs: gi = g@W_ih^T + b_ih, gh = h@W_hh^T + b_hh ------------
// Outputs stored fragment-packed: gi_p[((tile*24+nt)*32 + lane)*8 + r].

__global__ void gru_gemm(const __bf16* __restrict__ gbf,
                         const __bf16* __restrict__ hbf,
                         const __bf16* __restrict__ wih_p,
                         const __bf16* __restrict__ whh_p,
                         const float* __restrict__ bih,
                         const float* __restrict__ bhh,
                         float* __restrict__ gi, float* __restrict__ gh) {
    const int wave = threadIdx.x >> 5;
    const int lane = threadIdx.x & 31;
    const int tile = blockIdx.x * 4 + wave;
    if (tile * 16 >= N_NODES) return;                   // wave-uniform exit
    const int m0 = tile * 16;
    const int lr = lane & 15;

    v16bf ag[4], ah[4];
#pragma unroll
    for (int ks = 0; ks < 4; ++ks) {
        ag[ks] = load_a_frag_bf16(gbf, D_DIM, m0, ks * 32, lane);
        ah[ks] = load_a_frag_bf16(hbf, D_DIM, m0, ks * 32, lane);
    }

#pragma unroll
    for (int nt = 0; nt < 24; ++nt) {                   // 384 output cols
        const int n0 = nt * 16;
        v8f ci = splat8(bih[n0 + lr]);
        v8f ch = splat8(bhh[n0 + lr]);
#pragma unroll
        for (int ks = 0; ks < 4; ++ks) {
            const v16bf b = load_b_packed(wih_p, nt * 4 + ks, lane);
            ci = __builtin_amdgcn_wmma_f32_16x16x32_bf16(
                false, ag[ks], false, b, (short)0, ci, false, false);
        }
#pragma unroll
        for (int ks = 0; ks < 4; ++ks) {
            const v16bf b = load_b_packed(whh_p, nt * 4 + ks, lane);
            ch = __builtin_amdgcn_wmma_f32_16x16x32_bf16(
                false, ah[ks], false, b, (short)0, ch, false, false);
        }
        const size_t fo = (((size_t)tile * 24 + nt) * 32 + lane) << 3;
        *(v8f*)(gi + fo) = ci;                          // 32B contiguous/lane
        *(v8f*)(gh + fo) = ch;
    }
}

// ---- GRU elementwise update (torch gate order r,z,n) -----------------------
// Reads the fragment-packed gi/gh layout via the inverse index mapping.

__global__ void gru_update(const float* __restrict__ gi,
                           const float* __restrict__ gh,
                           float* __restrict__ h, __bf16* __restrict__ hbf) {
    const int tid = blockIdx.x * blockDim.x + threadIdx.x;
    if (tid >= N_NODES * D_DIM) return;
    const int n = tid >> 7, j = tid & 127;
    const int tile  = n >> 4;
    const int rowin = n & 15;
    const int r     = rowin & 7;
    const int lane  = (rowin >> 3) * 16 + (j & 15);
    const int ntb   = j >> 4;
    const size_t i0 = ((((size_t)tile * 24 + ntb)      * 32 + lane) << 3) + r;
    const size_t i1 = ((((size_t)tile * 24 + ntb + 8)  * 32 + lane) << 3) + r;
    const size_t i2 = ((((size_t)tile * 24 + ntb + 16) * 32 + lane) << 3) + r;
    const float ir = gi[i0], hr = gh[i0];
    const float iz = gi[i1], hz = gh[i1];
    const float in_ = gi[i2], hn = gh[i2];
    const float hp = h[tid];
    const float rr = 1.f / (1.f + __expf(-(ir + hr)));
    const float zz = 1.f / (1.f + __expf(-(iz + hz)));
    const float nn = tanhf(in_ + rr * hn);
    const float hv = (1.f - zz) * nn + zz * hp;
    h[tid]   = hv;
    hbf[tid] = (__bf16)hv;
}

// ---- final readout: out[n] = h[n] . W_lin + b_lin --------------------------

__global__ void final_linear(const float* __restrict__ h,
                             const float* __restrict__ wlin,
                             const float* __restrict__ blin,
                             float* __restrict__ out) {
    const int n = blockIdx.x * blockDim.x + threadIdx.x;
    if (n >= N_NODES) return;
    const float4* hp = (const float4*)(h + (size_t)n * D_DIM);
    const float4* wp = (const float4*)wlin;
    float s = 0.f;
#pragma unroll
    for (int i = 0; i < 32; ++i) {
        const float4 a = hp[i], b = wp[i];
        s += a.x * b.x + a.y * b.y + a.z * b.z + a.w * b.w;
    }
    out[n] = s + blin[0];
}

// ---------------------------------------------------------------------------

extern "C" void kernel_launch(void* const* d_in, const int* in_sizes, int n_in,
                              void* d_out, int out_size, void* d_ws, size_t ws_size,
                              hipStream_t stream) {
    const float* x     = (const float*)d_in[0];   // [N,T,128]
    const int*   ei    = (const int*)  d_in[1];   // [T,2,E]
    const float* Wgat  = (const float*)d_in[2];   // [128,128]
    const float* a_src = (const float*)d_in[3];   // [4,32]
    const float* a_dst = (const float*)d_in[4];
    const float* b_gat = (const float*)d_in[5];   // [128]
    const float* W_ih  = (const float*)d_in[6];   // [384,128]
    const float* W_hh  = (const float*)d_in[7];
    const float* b_ih  = (const float*)d_in[8];   // [384]
    const float* b_hh  = (const float*)d_in[9];
    const float* W_lin = (const float*)d_in[10];  // [1,128]
    const float* b_lin = (const float*)d_in[11];
    float* out = (float*)d_out;

    // workspace carve-up (~273 MB)
    char* ws = (char*)d_ws;
    size_t off = 0;
    auto take = [&](size_t bytes) -> char* {
        char* p = ws + off;
        off = (off + bytes + 255) & ~(size_t)255;
        return p;
    };
    __bf16* wgat_p = (__bf16*)take((size_t)128 * 128 * 2);   // 32 frags
    __bf16* wih_p  = (__bf16*)take((size_t)128 * 384 * 2);   // 96 frags
    __bf16* whh_p  = (__bf16*)take((size_t)128 * 384 * 2);
    float*  hgat   = (float*) take((size_t)N_NODES * D_DIM * 4);
    float*  alsrc  = (float*) take((size_t)N_NODES * HEADS * 4);
    float*  aldst  = (float*) take((size_t)N_NODES * HEADS * 4);
    float*  mmax   = (float*) take((size_t)N_NODES * HEADS * 4);
    float*  denom  = (float*) take((size_t)N_NODES * HEADS * 4);
    float*  elog   = (float*) take((size_t)E_TOT * HEADS * 4);
    float*  outacc = (float*) take((size_t)N_NODES * D_DIM * 4);
    __bf16* gbf    = (__bf16*)take((size_t)N_NODES * D_DIM * 2);
    float*  hstate = (float*) take((size_t)N_NODES * D_DIM * 4);
    __bf16* hbf    = (__bf16*)take((size_t)N_NODES * D_DIM * 2);
    float*  gi     = (float*) take((size_t)N_NODES * 384 * 4);
    float*  gh     = (float*) take((size_t)N_NODES * 384 * 4);

    const int tiles      = N_NODES / 16;                        // 3125
    const int gemm_grid  = (tiles + 3) / 4;                     // 4 waves/block
    const int nh_grid    = (N_NODES * HEADS + 255) / 256;
    const int edge_grid  = (E_TOT + 255) / 256;
    const int emsg_grid  = (E_TOT + 7) / 8;                     // wave per edge
    const int nd_grid    = (N_NODES * D_DIM + 255) / 256;

    pack_b<<<(4 * 8 * 512 + 255) / 256, 256, 0, stream>>>(Wgat, wgat_p, 4, 8, 128, 0);
    pack_b<<<(4 * 24 * 512 + 255) / 256, 256, 0, stream>>>(W_ih, wih_p, 4, 24, 128, 1);
    pack_b<<<(4 * 24 * 512 + 255) / 256, 256, 0, stream>>>(W_hh, whh_p, 4, 24, 128, 1);

    hipMemsetAsync(hstate, 0, (size_t)N_NODES * D_DIM * 4, stream);
    hipMemsetAsync(hbf,    0, (size_t)N_NODES * D_DIM * 2, stream);

    for (int t = 0; t < T_STEPS; ++t) {
        gat_project<<<gemm_grid, 128, 0, stream>>>(x, wgat_p, hgat, t);
        compute_al<<<nh_grid, 256, 0, stream>>>(hgat, a_src, a_dst,
                                                alsrc, aldst, mmax, denom);
        hipMemsetAsync(outacc, 0, (size_t)N_NODES * D_DIM * 4, stream);
        edge_logits<<<edge_grid, 256, 0, stream>>>(ei, alsrc, aldst, elog, mmax, t);
        edge_expsum<<<edge_grid, 256, 0, stream>>>(ei, elog, mmax, denom, t);
        edge_message<<<emsg_grid, 256, 0, stream>>>(ei, elog, denom, hgat, outacc, t);
        gat_finalize<<<nd_grid, 256, 0, stream>>>(outacc, b_gat, gbf);
        gru_gemm<<<gemm_grid, 128, 0, stream>>>(gbf, hbf, wih_p, whh_p,
                                                b_ih, b_hh, gi, gh);
        gru_update<<<nd_grid, 256, 0, stream>>>(gi, gh, hstate, hbf);
    }

    final_linear<<<(N_NODES + 255) / 256, 256, 0, stream>>>(hstate, W_lin, b_lin, out);
}
